// RTERModel_10634339025021
// MI455X (gfx1250) — matrix operand order repaired and checked by reading, not compiled, not run on hardware
//
#include <hip/hip_runtime.h>
#include <hip/hip_bf16.h>
#include <math.h>

// ---------------- model dims ----------------
#define B_UTT 256
#define T_TOKN 80
#define IN_DIM 300
#define IN_PAD 320
#define HID 512
#define G3 1536          // 3*HID
#define WIN 40
#define NQ 255           // N_UTT - 1
#define NCLS 7
#define CHNK 10          // timestep chunk for batched input GEMMs

typedef __bf16 bf16;
typedef __attribute__((ext_vector_type(16))) __bf16 v16bf;
typedef __attribute__((ext_vector_type(8)))  __bf16 v8bf;
typedef __attribute__((ext_vector_type(8)))  float  v8f;

#if defined(__HIP_DEVICE_COMPILE__) && defined(__gfx1250__) && \
    __has_builtin(__builtin_amdgcn_global_load_async_to_lds_b128) && \
    __has_builtin(__builtin_amdgcn_s_wait_asynccnt)
#define USE_ASYNC_LDS 1
#else
#define USE_ASYNC_LDS 0
#endif

// Universal pointer adapter: converts to whatever pointer type (incl. vector
// pointee and address-space qualifiers) a builtin's prototype demands.
struct any_ptr_ {
  const void* p;
  template <typename T>
  __device__ __forceinline__ operator T*() const { return (T*)p; }
};

__device__ __forceinline__ float sigmoidf_(float x){ return 1.f/(1.f + __expf(-x)); }

// ---------------- WMMA GEMM:  C[M,N] = A[M,K](bf16,row) @ W[N,K](bf16,row)^T (+bias[N]) ----------------
// block = 128 threads (4 waves); block tile 64(M) x 64(N); wave tile 16 x 64.
// B panel staged to LDS in 64x32 chunks (4KB), double buffered, async when available.
// A fragment (16x32 bf16): lane L holds row (L&15); lanes<16: K k..k+15; lanes>=16: K k+16..k+31.
// B fragment (32x16 bf16): lane L holds col (L&15); lanes<16: K k..k+15; lanes>=16: K k+16..k+31.
__device__ __forceinline__ v16bf load_a_frag(const bf16* __restrict__ A, int lda, int m0, int k0, int lane){
  int m = m0 + (lane & 15);
  int s = (lane & 16) ? 16 : 0;
  const bf16* base = A + (size_t)m * lda + k0 + s;
  v8bf lo = *(const v8bf*)(base);
  v8bf hi = *(const v8bf*)(base + 8);
  v16bf f;
#pragma unroll
  for (int i = 0; i < 8; ++i){ f[i] = lo[i]; f[i+8] = hi[i]; }
  return f;
}

// staged chunk layout: col n (0..63) at lbuf + n*64, 32 bf16 (64B) of K values
__device__ __forceinline__ v16bf load_b_lds(const char* lbuf, int nt, int lane){
  int n  = nt * 16 + (lane & 15);
  int ko = (lane & 16) ? 32 : 0;   // byte offset of K half
  return *(const v16bf*)(lbuf + n * 64 + ko);
}

__device__ __forceinline__ void stage_b_16(const bf16* __restrict__ g, char* l){
#if USE_ASYNC_LDS
  __builtin_amdgcn_global_load_async_to_lds_b128(any_ptr_{g}, any_ptr_{l}, 0, 0);
#else
  *(uint4*)l = *(const uint4*)g;
#endif
}

__device__ __forceinline__ void stage_b(const bf16* __restrict__ Wt, int K, int n0, int kc,
                                        char* lbuf, int tid){
  int n = tid >> 2;           // 0..31
  int part = tid & 3;         // 16B piece within 64B row
  const bf16* g0 = Wt + (size_t)(n0 + n) * K + kc + part * 8;
  const bf16* g1 = Wt + (size_t)(n0 + n + 32) * K + kc + part * 8;
  stage_b_16(g0, lbuf + n * 64 + part * 16);
  stage_b_16(g1, lbuf + (n + 32) * 64 + part * 16);
}

__device__ __forceinline__ void async_fence_(){
#if USE_ASYNC_LDS
  __builtin_amdgcn_s_wait_asynccnt(0);
#endif
  __syncthreads();
}

__global__ void __launch_bounds__(128)
gemm_bf16nt(const bf16* __restrict__ A, const bf16* __restrict__ Wt,
            float* __restrict__ C, const float* __restrict__ bias,
            int N, int K){
  __shared__ __align__(32) char ldsB[2][64 * 64];
  const int tid  = threadIdx.x;
  const int lane = tid & 31;
  const int wave = tid >> 5;
  const int m0 = blockIdx.x * 64 + wave * 16;
  const int n0 = blockIdx.y * 64;

  stage_b(Wt, K, n0, 0, ldsB[0], tid);
  async_fence_();

  v8f acc0 = {}; v8f acc1 = {}; v8f acc2 = {}; v8f acc3 = {};
  int buf = 0;
  for (int k = 0; k < K; k += 32){
    const bool more = (k + 32) < K;
    if (more) stage_b(Wt, K, n0, k + 32, ldsB[buf ^ 1], tid);   // prefetch overlaps WMMA
    v16bf a  = load_a_frag(A, K, m0, k, lane);
    v16bf b0 = load_b_lds(ldsB[buf], 0, lane);
    v16bf b1 = load_b_lds(ldsB[buf], 1, lane);
    v16bf b2 = load_b_lds(ldsB[buf], 2, lane);
    v16bf b3 = load_b_lds(ldsB[buf], 3, lane);
    acc0 = __builtin_amdgcn_wmma_f32_16x16x32_bf16(false, a, false, b0, (short)0, acc0, false, false);
    acc1 = __builtin_amdgcn_wmma_f32_16x16x32_bf16(false, a, false, b1, (short)0, acc1, false, false);
    acc2 = __builtin_amdgcn_wmma_f32_16x16x32_bf16(false, a, false, b2, (short)0, acc2, false, false);
    acc3 = __builtin_amdgcn_wmma_f32_16x16x32_bf16(false, a, false, b3, (short)0, acc3, false, false);
    if (more){ async_fence_(); buf ^= 1; }
  }
  const int nn = lane & 15;
  const int mo = (lane & 16) ? 8 : 0;
  float bv0 = bias ? bias[n0 + nn +  0] : 0.f;
  float bv1 = bias ? bias[n0 + nn + 16] : 0.f;
  float bv2 = bias ? bias[n0 + nn + 32] : 0.f;
  float bv3 = bias ? bias[n0 + nn + 48] : 0.f;
#pragma unroll
  for (int r = 0; r < 8; ++r){
    int row = m0 + mo + r;
    float* cp = C + (size_t)row * N + n0 + nn;
    cp[ 0] = acc0[r] + bv0;
    cp[16] = acc1[r] + bv1;
    cp[32] = acc2[r] + bv2;
    cp[48] = acc3[r] + bv3;
  }
}

// ---------------- elementwise / small kernels ----------------
__global__ void cvt_pad(const float* __restrict__ src, bf16* __restrict__ dst,
                        int rows, int kin, int kout){
  int i = blockIdx.x * 256 + threadIdx.x;
  if (i >= rows * kout) return;
  int r = i / kout, k = i - r * kout;
  dst[i] = (k < kin) ? (bf16)src[(size_t)r * kin + k] : (bf16)0.f;
}

__global__ void embed_gather(const int* __restrict__ ids, const float* __restrict__ emb,
                             bf16* __restrict__ X){
  int row = blockIdx.x;          // t*B + b
  int t = row >> 8;
  int b = row & 255;
  int id = ids[(size_t)b * T_TOKN + t];
  int j = threadIdx.x;           // 320 threads
  X[(size_t)row * IN_PAD + j] = (j < IN_DIM) ? (bf16)emb[(size_t)id * IN_DIM + j] : (bf16)0.f;
}

__global__ void gru_gate(const float* __restrict__ gi, const float* __restrict__ gh,
                         const float* __restrict__ bih, const float* __restrict__ bhh,
                         float* __restrict__ h, bf16* __restrict__ hbf,
                         bf16* __restrict__ out_bf, float* __restrict__ out_f32,
                         const int* __restrict__ lens, int t){
  int i = blockIdx.x * 256 + threadIdx.x;   // B*HID
  int b = i >> 9, j = i & 511;
  size_t o = (size_t)b * G3 + j;
  float ir  = gi[o]            + bih[j];
  float iz  = gi[o +   HID]    + bih[HID + j];
  float in_ = gi[o + 2*HID]    + bih[2*HID + j];
  float hr  = gh[o]            + bhh[j];
  float hz  = gh[o +   HID]    + bhh[HID + j];
  float hn  = gh[o + 2*HID]    + bhh[2*HID + j];
  float r  = sigmoidf_(ir + hr);
  float z  = sigmoidf_(iz + hz);
  float nn = tanhf(in_ + r * hn);
  float hp = h[i];
  float hnew = (1.f - z) * nn + z * hp;
  bool valid = lens ? (t < lens[b]) : true;
  float hc = valid ? hnew : hp;
  h[i] = hc;
  hbf[i] = (bf16)hc;
  if (out_bf)  out_bf[i]  = (bf16)(valid ? hnew : 0.f);
  if (out_f32) out_f32[i] = valid ? hnew : 0.f;
}

__global__ void att_score(const bf16* __restrict__ outF, const bf16* __restrict__ outB,
                          const float* __restrict__ apw, const float* __restrict__ apb,
                          float* __restrict__ s){
  int t = blockIdx.x, n = blockIdx.y;
  const bf16* f = outF + ((size_t)t * B_UTT + n) * HID;
  const bf16* b = outB + ((size_t)t * B_UTT + n) * HID;
  float acc = 0.f;
  for (int j = threadIdx.x; j < HID; j += 256)
    acc += (float)f[j] * apw[j] + (float)b[j] * apw[HID + j];
  __shared__ float sm[256];
  sm[threadIdx.x] = acc; __syncthreads();
  for (int st = 128; st > 0; st >>= 1){
    if (threadIdx.x < st) sm[threadIdx.x] += sm[threadIdx.x + st];
    __syncthreads();
  }
  if (threadIdx.x == 0) s[(size_t)n * T_TOKN + t] = sm[0] + apb[0];
}

__global__ void softmax_T(float* __restrict__ s){
  int n = blockIdx.x, tid = threadIdx.x;       // 128 threads
  float* row = s + (size_t)n * T_TOKN;
  float v = (tid < T_TOKN) ? row[tid] : -3.0e38f;
  __shared__ float sm[128];
  sm[tid] = v; __syncthreads();
  for (int st = 64; st > 0; st >>= 1){ if (tid < st) sm[tid] = fmaxf(sm[tid], sm[tid + st]); __syncthreads(); }
  float mx = sm[0]; __syncthreads();
  float e = (tid < T_TOKN) ? __expf(v - mx) : 0.f;
  sm[tid] = e; __syncthreads();
  for (int st = 64; st > 0; st >>= 1){ if (tid < st) sm[tid] += sm[tid + st]; __syncthreads(); }
  if (tid < T_TOKN) row[tid] = e / sm[0];
}

__global__ void pool_kernel(const bf16* __restrict__ outF, const bf16* __restrict__ outB,
                            const float* __restrict__ attw,
                            float* __restrict__ pool1, float* __restrict__ pool2){
  int n = blockIdx.x;
  for (int d = threadIdx.x; d < 2 * HID; d += 256){
    float wsum = 0.f, mx = -3.0e38f;
    for (int t = 0; t < T_TOKN; ++t){
      float v = (d < HID) ? (float)outF[((size_t)t * B_UTT + n) * HID + d]
                          : (float)outB[((size_t)t * B_UTT + n) * HID + d - HID];
      wsum += attw[(size_t)n * T_TOKN + t] * v;
      mx = fmaxf(mx, v);
    }
    pool1[(size_t)n * 2 * HID + d] = wsum;
    pool2[(size_t)n * 2 * HID + d] = mx;
  }
}

__global__ void compute_u(const float* __restrict__ p1, const float* __restrict__ p2,
                          const float* __restrict__ l1w, const float* __restrict__ l1b,
                          const float* __restrict__ l2w, const float* __restrict__ l2b,
                          float* __restrict__ u){
  int i = blockIdx.x * 256 + threadIdx.x;      // B*HID
  int n = i >> 9, j = i & 511;
  const float* r1 = p1 + (size_t)n * 1024;
  const float* r2 = p2 + (size_t)n * 1024;
  const float* w1 = l1w + (size_t)j * 1024;
  const float* w2 = l2w + (size_t)j * 1024;
  float a1 = l1b[j], a2 = l2b[j];
  for (int k = 0; k < 1024; ++k){ a1 += r1[k] * w1[k]; a2 += r2[k] * w2[k]; }
  u[i] = tanhf(a1 + a2);
}

__global__ void build_batches(const float* __restrict__ u, bf16* __restrict__ bat){
  int q = blockIdx.x, w = blockIdx.y;          // grid (256, WIN), block HID
  int j = threadIdx.x;
  int idx = (q + 1) - WIN + w;
  float v = (q < NQ && idx >= 0) ? u[(size_t)idx * HID + j] : 0.f;
  bat[((size_t)w * B_UTT + q) * HID + j] = (bf16)v;
}

__global__ void make_memb(bf16* __restrict__ bat, const float* __restrict__ ff,
                          const float* __restrict__ fb){
  int i = blockIdx.x * 256 + threadIdx.x;      // WIN*B*HID
  bat[i] = (bf16)((float)bat[i] + ff[i] + fb[i]);
}

__global__ void qw_kernel(const float* __restrict__ u, const float* __restrict__ agW,
                          const float* __restrict__ agWb, float* __restrict__ qW){
  int i = blockIdx.x * 256 + threadIdx.x;      // B*HID
  int q = i >> 9, h = i & 511;
  if (q >= NQ){ qW[i] = 0.f; return; }
  const float* uv = u + (size_t)(q + 1) * HID;
  const float* wv = agW + (size_t)h * HID;
  float a = agWb[h];
  for (int k = 0; k < HID; ++k) a += uv[k] * wv[k];
  qW[i] = a;
}

__global__ void vv_kernel(const float* __restrict__ qW, const float* __restrict__ agU,
                          float* __restrict__ vv){
  int i = blockIdx.x * 256 + threadIdx.x;      // B*HID
  int q = i >> 9, k = i & 511;
  if (q >= NQ){ vv[i] = 0.f; return; }
  const float* qv = qW + (size_t)q * HID;
  float a = 0.f;
  for (int h = 0; h < HID; ++h) a += qv[h] * agU[(size_t)h * HID + k];
  vv[i] = a;
}

__global__ void qb_kernel(const float* __restrict__ qW, const float* __restrict__ agUb,
                          float* __restrict__ qb){
  int q = threadIdx.x;                          // 256 threads
  float a = 0.f;
  if (q < NQ){ for (int h = 0; h < HID; ++h) a += qW[(size_t)q * HID + h] * agUb[h]; }
  qb[q] = a;
}

__global__ void sc_kernel(const float* __restrict__ vv, const bf16* __restrict__ memb,
                          const float* __restrict__ qb, float* __restrict__ sc){
  int q = blockIdx.x, w = blockIdx.y;          // grid (NQ, WIN)
  const float* vq = vv + (size_t)q * HID;
  const bf16* mm = memb + ((size_t)w * B_UTT + q) * HID;
  float acc = 0.f;
  for (int k = threadIdx.x; k < HID; k += 256) acc += vq[k] * (float)mm[k];
  __shared__ float sm[256];
  sm[threadIdx.x] = acc; __syncthreads();
  for (int st = 128; st > 0; st >>= 1){
    if (threadIdx.x < st) sm[threadIdx.x] += sm[threadIdx.x + st];
    __syncthreads();
  }
  if (threadIdx.x == 0) sc[(size_t)q * WIN + w] = sm[0] + qb[q];
}

__global__ void gates_kernel(const float* __restrict__ sc, float* __restrict__ gates){
  int q = blockIdx.x, w = threadIdx.x;         // grid 256, block 64
  if (q >= NQ){ if (w < WIN) gates[(size_t)q * WIN + w] = 0.f; return; }
  __shared__ float sm[64];
  bool valid = (w < WIN) && ((q + 1 - WIN + w) >= 0);
  float x = valid ? sc[(size_t)q * WIN + w] : -3.0e38f;
  sm[w] = x; __syncthreads();
  for (int st = 32; st > 0; st >>= 1){ if (w < st) sm[w] = fmaxf(sm[w], sm[w + st]); __syncthreads(); }
  float mx = sm[0]; __syncthreads();
  float e = valid ? __expf(x - mx) : 0.f;
  sm[w] = e; __syncthreads();
  for (int st = 32; st > 0; st >>= 1){ if (w < st) sm[w] += sm[w + st]; __syncthreads(); }
  if (w < WIN) gates[(size_t)q * WIN + w] = e / sm[0];
}

__global__ void att_cell(const float* __restrict__ xr, const float* __restrict__ xn,
                         const float* __restrict__ ghr, const float* __restrict__ ghn,
                         const float* __restrict__ bhr, const float* __restrict__ bhn,
                         const float* __restrict__ gates, float* __restrict__ h,
                         bf16* __restrict__ hbf, int w){
  int i = blockIdx.x * 256 + threadIdx.x;      // B*HID
  int q = i >> 9, j = i & 511;
  float r  = sigmoidf_(xr[i] + ghr[i] + bhr[j]);
  float nn = tanhf(xn[i] + r * (ghn[i] + bhn[j]));
  float g  = gates[(size_t)q * WIN + w];
  float hp = h[i];
  float hnew = g * nn + (1.f - g) * hp;
  h[i] = hnew;
  hbf[i] = (bf16)hnew;
}

__global__ void cls_kernel(const float* __restrict__ u, const float* __restrict__ hf,
                           const float* __restrict__ hb, const float* __restrict__ clsw,
                           const float* __restrict__ clsb, float* __restrict__ out){
  int n = blockIdx.x, c = threadIdx.x;         // grid 256, block 32
  if (c >= NCLS) return;
  const float* wv = clsw + (size_t)c * HID;
  float a = clsb[c];
  for (int j = 0; j < HID; ++j){
    float f = (n == 0) ? u[j]
                       : (u[(size_t)n * HID + j] + hf[(size_t)(n - 1) * HID + j] + hb[(size_t)(n - 1) * HID + j]);
    a += f * wv[j];
  }
  out[(size_t)n * NCLS + c] = a;
}

// ---------------- host orchestration ----------------
extern "C" void kernel_launch(void* const* d_in, const int* in_sizes, int n_in,
                              void* d_out, int out_size, void* d_ws, size_t ws_size,
                              hipStream_t stream) {
  (void)in_sizes; (void)n_in; (void)out_size; (void)ws_size;
  const int*   ids    = (const int*)  d_in[0];
  const int*   lens   = (const int*)  d_in[1];
  const float* emb    = (const float*)d_in[2];
  const float* uttWih = (const float*)d_in[3];
  const float* uttWhh = (const float*)d_in[4];
  const float* uttbih = (const float*)d_in[5];
  const float* uttbhh = (const float*)d_in[6];
  const float* apw    = (const float*)d_in[7];
  const float* apb    = (const float*)d_in[8];
  const float* l1w    = (const float*)d_in[9];
  const float* l1b    = (const float*)d_in[10];
  const float* l2w    = (const float*)d_in[11];
  const float* l2b    = (const float*)d_in[12];
  const float* fusWih = (const float*)d_in[13];
  const float* fusWhh = (const float*)d_in[14];
  const float* fusbih = (const float*)d_in[15];
  const float* fusbhh = (const float*)d_in[16];
  const float* agW    = (const float*)d_in[17];
  const float* agWb   = (const float*)d_in[18];
  const float* agU    = (const float*)d_in[19];
  const float* agUb   = (const float*)d_in[20];
  const float* cWir   = (const float*)d_in[21];
  const float* cbir   = (const float*)d_in[22];
  const float* cWhr   = (const float*)d_in[23];
  const float* cbhr   = (const float*)d_in[24];
  const float* cWin   = (const float*)d_in[25];
  const float* cbin   = (const float*)d_in[26];
  const float* cWhn   = (const float*)d_in[27];
  const float* cbhn   = (const float*)d_in[28];
  const float* clsw   = (const float*)d_in[29];
  const float* clsb   = (const float*)d_in[30];

  char* ws = (char*)d_ws;
  size_t off = 0;
  auto take = [&](size_t bytes) -> char* {
    char* p = ws + off;
    off += (bytes + 255) & ~(size_t)255;
    return p;
  };

  // bf16 weights
  bf16* w_uttWih = (bf16*)take((size_t)2 * G3 * IN_PAD * 2);
  bf16* w_uttWhh = (bf16*)take((size_t)2 * G3 * HID * 2);
  bf16* w_fusWih = (bf16*)take((size_t)2 * G3 * HID * 2);
  bf16* w_fusWhh = (bf16*)take((size_t)2 * G3 * HID * 2);
  bf16* w_cWir   = (bf16*)take((size_t)2 * HID * HID * 2);
  bf16* w_cWhr   = (bf16*)take((size_t)2 * HID * HID * 2);
  bf16* w_cWin   = (bf16*)take((size_t)2 * HID * HID * 2);
  bf16* w_cWhn   = (bf16*)take((size_t)2 * HID * HID * 2);
  // small f32 state
  float* hbuf  = (float*)take((size_t)B_UTT * HID * 4);
  bf16*  hbf   = (bf16*) take((size_t)B_UTT * HID * 2);
  float* giChunk = (float*)take((size_t)CHNK * B_UTT * G3 * 4);   // 15.7 MB chunked input gates
  float* ghbuf = (float*)take((size_t)B_UTT * G3 * 4);
  float* grbuf = (float*)take((size_t)B_UTT * G3 * 4);            // ghr/ghn for att cells
  float* ubuf  = (float*)take((size_t)B_UTT * HID * 4);
  float* pool1 = (float*)take((size_t)B_UTT * 1024 * 4);
  float* pool2 = (float*)take((size_t)B_UTT * 1024 * 4);
  float* attw  = (float*)take((size_t)B_UTT * T_TOKN * 4);
  float* qWb   = (float*)take((size_t)B_UTT * HID * 4);
  float* vvb   = (float*)take((size_t)B_UTT * HID * 4);
  float* qbb   = (float*)take(1024);
  float* scb   = (float*)take((size_t)B_UTT * WIN * 4);
  float* gatesb= (float*)take((size_t)B_UTT * WIN * 4);
  float* hfb   = (float*)take((size_t)B_UTT * HID * 4);
  float* hbb   = (float*)take((size_t)B_UTT * HID * 4);
  // big reused region A
  char* regA = take((size_t)56 * 1024 * 1024);
  // phase 2 layout
  bf16* Xemb = (bf16*)regA;                                      // T*B*IN_PAD bf16 = 13,107,200 B
  bf16* outF = (bf16*)(regA + 13107200);                         // T*B*HID bf16  = 20,971,520 B
  bf16* outB = (bf16*)(regA + 13107200 + 20971520);
  // phase 4+ layout (reuse)
  bf16*  bat = (bf16*)regA;                                      // WIN*B*HID bf16 = 10,485,760 B
  float* ffb = (float*)(regA + 10485760);                        // WIN*B*HID f32  = 20,971,520 B
  float* fbb = (float*)(regA + 10485760 + 20971520);
  float* xrb = ffb;
  float* xnb = fbb;

  auto gemm = [&](const bf16* A, const bf16* Wt, float* C, const float* bias,
                  int M, int N, int K){
    gemm_bf16nt<<<dim3(M / 64, N / 64), 128, 0, stream>>>(A, Wt, C, bias, N, K);
  };
  auto cvt = [&](const float* s, bf16* d, int rows, int kin, int kout){
    int tot = rows * kout;
    cvt_pad<<<(tot + 255) / 256, 256, 0, stream>>>(s, d, rows, kin, kout);
  };

  // ---- 1. weights to bf16 (Wih padded 300 -> 320) ----
  cvt(uttWih, w_uttWih, 2 * G3, IN_DIM, IN_PAD);
  cvt(uttWhh, w_uttWhh, 2 * G3, HID, HID);
  cvt(fusWih, w_fusWih, 2 * G3, HID, HID);
  cvt(fusWhh, w_fusWhh, 2 * G3, HID, HID);
  cvt(cWir, w_cWir, 2 * HID, HID, HID);
  cvt(cWhr, w_cWhr, 2 * HID, HID, HID);
  cvt(cWin, w_cWin, 2 * HID, HID, HID);
  cvt(cWhn, w_cWhn, 2 * HID, HID, HID);

  // ---- 2. embedding gather (padded bf16) ----
  embed_gather<<<T_TOKN * B_UTT, IN_PAD, 0, stream>>>(ids, emb, Xemb);

  // ---- 3. utterance BiGRU over tokens (input GEMMs batched per 10-step chunk) ----
  for (int d = 0; d < 2; ++d){
    (void)hipMemsetAsync(hbuf, 0, (size_t)B_UTT * HID * 4, stream);
    (void)hipMemsetAsync(hbf,  0, (size_t)B_UTT * HID * 2, stream);
    const int nch = T_TOKN / CHNK;
    for (int ci = 0; ci < nch; ++ci){
      int c = (d == 0) ? ci : (nch - 1 - ci);
      gemm(Xemb + (size_t)c * CHNK * B_UTT * IN_PAD, w_uttWih + (size_t)d * G3 * IN_PAD,
           giChunk, nullptr, CHNK * B_UTT, G3, IN_PAD);
      for (int si = 0; si < CHNK; ++si){
        int tt = (d == 0) ? si : (CHNK - 1 - si);
        int t = c * CHNK + tt;
        gemm(hbf, w_uttWhh + (size_t)d * G3 * HID, ghbuf, nullptr, B_UTT, G3, HID);
        gru_gate<<<(B_UTT * HID) / 256, 256, 0, stream>>>(
            giChunk + (size_t)tt * B_UTT * G3, ghbuf,
            uttbih + (size_t)d * G3, uttbhh + (size_t)d * G3,
            hbuf, hbf, (d == 0 ? outF : outB) + (size_t)t * B_UTT * HID, nullptr, lens, t);
      }
    }
  }

  // ---- 4. attention + max pooling -> u ----
  att_score<<<dim3(T_TOKN, B_UTT), 256, 0, stream>>>(outF, outB, apw, apb, attw);
  softmax_T<<<B_UTT, 128, 0, stream>>>(attw);
  pool_kernel<<<B_UTT, 256, 0, stream>>>(outF, outB, attw, pool1, pool2);
  compute_u<<<(B_UTT * HID) / 256, 256, 0, stream>>>(pool1, pool2, l1w, l1b, l2w, l2b, ubuf);

  // ---- 5. sliding-window memory (bf16) ----
  build_batches<<<dim3(B_UTT, WIN), HID, 0, stream>>>(ubuf, bat);

  // ---- 6. fusion BiGRU over window (input GEMMs batched per 10-step chunk) ----
  for (int d = 0; d < 2; ++d){
    (void)hipMemsetAsync(hbuf, 0, (size_t)B_UTT * HID * 4, stream);
    (void)hipMemsetAsync(hbf,  0, (size_t)B_UTT * HID * 2, stream);
    const int nch = WIN / CHNK;
    for (int ci = 0; ci < nch; ++ci){
      int c = (d == 0) ? ci : (nch - 1 - ci);
      gemm(bat + (size_t)c * CHNK * B_UTT * HID, w_fusWih + (size_t)d * G3 * HID,
           giChunk, nullptr, CHNK * B_UTT, G3, HID);
      for (int si = 0; si < CHNK; ++si){
        int tt = (d == 0) ? si : (CHNK - 1 - si);
        int w = c * CHNK + tt;
        gemm(hbf, w_fusWhh + (size_t)d * G3 * HID, ghbuf, nullptr, B_UTT, G3, HID);
        gru_gate<<<(B_UTT * HID) / 256, 256, 0, stream>>>(
            giChunk + (size_t)tt * B_UTT * G3, ghbuf,
            fusbih + (size_t)d * G3, fusbhh + (size_t)d * G3,
            hbuf, hbf, nullptr, (d == 0 ? ffb : fbb) + (size_t)w * B_UTT * HID, nullptr, w);
      }
    }
  }

  // ---- 7. mem_bank = batches + ff + fb (in place, bf16) ----
  make_memb<<<(WIN * B_UTT * HID) / 256, 256, 0, stream>>>(bat, ffb, fbb);

  // ---- 8. agreement attention gates ----
  qw_kernel<<<(B_UTT * HID) / 256, 256, 0, stream>>>(ubuf, agW, agWb, qWb);
  vv_kernel<<<(B_UTT * HID) / 256, 256, 0, stream>>>(qWb, agU, vvb);
  qb_kernel<<<1, 256, 0, stream>>>(qWb, agUb, qbb);
  sc_kernel<<<dim3(NQ, WIN), 256, 0, stream>>>(vvb, bat, qbb, scb);
  gates_kernel<<<B_UTT, 64, 0, stream>>>(scb, gatesb);

  // ---- 9. attention GRU cells (fwd then bwd) ----
  for (int d = 0; d < 2; ++d){
    float* hH = (d == 0) ? hfb : hbb;
    gemm(bat, w_cWir + (size_t)d * HID * HID, xrb, cbir + (size_t)d * HID,
         WIN * B_UTT, HID, HID);
    gemm(bat, w_cWin + (size_t)d * HID * HID, xnb, cbin + (size_t)d * HID,
         WIN * B_UTT, HID, HID);
    (void)hipMemsetAsync(hH,  0, (size_t)B_UTT * HID * 4, stream);
    (void)hipMemsetAsync(hbf, 0, (size_t)B_UTT * HID * 2, stream);
    for (int s = 0; s < WIN; ++s){
      int w = (d == 0) ? s : (WIN - 1 - s);
      gemm(hbf, w_cWhr + (size_t)d * HID * HID, ghbuf, nullptr, B_UTT, HID, HID);
      gemm(hbf, w_cWhn + (size_t)d * HID * HID, grbuf, nullptr, B_UTT, HID, HID);
      att_cell<<<(B_UTT * HID) / 256, 256, 0, stream>>>(
          xrb + (size_t)w * B_UTT * HID, xnb + (size_t)w * B_UTT * HID,
          ghbuf, grbuf, cbhr + (size_t)d * HID, cbhn + (size_t)d * HID,
          gatesb, hH, hbf, w);
    }
  }

  // ---- 10. classifier ----
  cls_kernel<<<B_UTT, 32, 0, stream>>>(ubuf, hfb, hbb, clsw, clsb, (float*)d_out);
}
